// COMPASSModelV2_17334488006972
// MI455X (gfx1250) — compile-verified, matrix operand order checked by reading, not run on hardware
//
#include <hip/hip_runtime.h>

#define NN 2048
#define EE 65536
#define HIDD 256
#define NHH 8
#define DHH 32
#define ETT 5
#define EDIMM 6
#define INDIM 64

typedef __bf16 bf16t;
typedef __attribute__((ext_vector_type(16))) __bf16 v16bf;
typedef __attribute__((ext_vector_type(8)))  float v8f;

// ---------- bf16 helpers (bit-level, RNE) ----------
__device__ __forceinline__ unsigned short f2bfu(float f) {
  union { float f; unsigned u; } c; c.f = f;
  unsigned r = (c.u + 0x7FFFu + ((c.u >> 16) & 1u)) >> 16;
  return (unsigned short)r;
}
__device__ __forceinline__ float bfu2f(unsigned short s) {
  union { unsigned u; float f; } c; c.u = ((unsigned)s) << 16; return c.f;
}
__device__ __forceinline__ bf16t u2b(unsigned short s) {
  union { unsigned short s; bf16t b; } c; c.s = s; return c.b;
}
// CDNA5 wave32 16-bit A/B fragment K-index for element e (0..15), half = lane>>4
__device__ __forceinline__ int a_kidx(int e, int half) {
  int v = e >> 1, p = e & 1;
  return ((v & 4) << 2) + half * 8 + ((v & 3) << 1) + p;
}

// =====================================================================
// Generic GEMM:  C[M,Nout](ldc) = act(A[M,K] @ W[Nout,K]^T + bias)
// bf16 WMMA, f32 accumulate, LDS-staged tiles, ragged K/M/N via guards.
// block = 256 threads = 8 waves (4x2) -> 64x64 block tile,
// each wave owns a 16x32 slab => 2 WMMAs per K-step reusing one A frag.
// =====================================================================
#define GT_M 64
#define GT_N 64
#define GT_K 32

__global__ __launch_bounds__(256)
void gemm_bf16_kernel(const float* __restrict__ A, const float* __restrict__ W,
                      const float* __restrict__ bias, float* __restrict__ C,
                      int M, int K, int Nout, int ldc, int relu_flag)
{
  __shared__ unsigned short sA[GT_M][GT_K + 2];
  __shared__ unsigned short sB[GT_N][GT_K + 2];
  const int tid  = threadIdx.x;
  const int wave = tid >> 5, lane = tid & 31;
  const int wm = wave & 3, wn = wave >> 2;          // wn in 0..1
  const int m0 = blockIdx.y * GT_M, n0 = blockIdx.x * GT_N;
  const int half = lane >> 4, l15 = lane & 15;
  v8f acc0 = {0.f,0.f,0.f,0.f,0.f,0.f,0.f,0.f};
  v8f acc1 = {0.f,0.f,0.f,0.f,0.f,0.f,0.f,0.f};

  for (int k0 = 0; k0 < K; k0 += GT_K) {
    for (int i = tid; i < GT_M * GT_K; i += 256) {
      int r = i >> 5, c = i & 31;
      int gm = m0 + r, gk = k0 + c;
      float v = 0.f;
      if (gm < M && gk < K) {
        v = A[(size_t)gm * K + gk];
        if (gk + GT_K < K) __builtin_prefetch(&A[(size_t)gm * K + gk + GT_K], 0, 1);
      }
      sA[r][c] = f2bfu(v);
    }
    for (int i = tid; i < GT_N * GT_K; i += 256) {
      int r = i >> 5, c = i & 31;
      int gn = n0 + r, gk = k0 + c;
      float v = 0.f;
      if (gn < Nout && gk < K) v = W[(size_t)gn * K + gk];
      sB[r][c] = f2bfu(v);
    }
    __syncthreads();
    v16bf af, bf0, bf1;
    #pragma unroll
    for (int e = 0; e < 16; e++) {
      int kk = a_kidx(e, half);
      af[e]  = u2b(sA[wm * 16 + l15][kk]);
      bf0[e] = u2b(sB[wn * 32 + l15][kk]);
      bf1[e] = u2b(sB[wn * 32 + 16 + l15][kk]);
    }
    acc0 = __builtin_amdgcn_wmma_f32_16x16x32_bf16(false, af, false, bf0,
                                                   (short)0, acc0, false, false);
    acc1 = __builtin_amdgcn_wmma_f32_16x16x32_bf16(false, af, false, bf1,
                                                   (short)0, acc1, false, false);
    __syncthreads();
  }
  #pragma unroll
  for (int r = 0; r < 8; r++) {
    int m = m0 + wm * 16 + r + 8 * half;
    if (m < M) {
      int n = n0 + wn * 32 + l15;
      if (n < Nout) {
        float v = acc0[r];
        if (bias) v += bias[n];
        if (relu_flag) v = fmaxf(v, 0.f);
        C[(size_t)m * ldc + n] = v;
      }
      n += 16;
      if (n < Nout) {
        float v = acc1[r];
        if (bias) v += bias[n];
        if (relu_flag) v = fmaxf(v, 0.f);
        C[(size_t)m * ldc + n] = v;
      }
    }
  }
}

// =====================================================================
// Flash attention, bf16 WMMA. One wave per (head, 16-query tile);
// all 8 waves of a block share one head, so K/V 32x32 bf16 tiles are
// staged cooperatively in LDS (8x reduction in global K/V traffic).
// QKV packed bf16 [nq][768]; optional bf16 bias [NH][nq][nk].
// O[nq][HID] gets head h at columns h*32..h*32+31 (pre out-proj).
// =====================================================================
__global__ __launch_bounds__(256)
void flash_attn_kernel(const unsigned short* __restrict__ QKV,
                       const unsigned short* __restrict__ bias,
                       float* __restrict__ O, int nq, int nk)
{
  const int ld = 3 * HIDD;
  const int tid  = threadIdx.x;
  const int wave = tid >> 5;
  const int lane = tid & 31;
  const int half = lane >> 4, l15 = lane & 15;
  const int h  = blockIdx.y;
  const int q0 = (blockIdx.x * 8 + wave) * 16;
  __shared__ unsigned short sK[32][34];      // [key_local][dh]
  __shared__ unsigned short sV[32][34];      // [key_local][dh]
  __shared__ unsigned short sP[8][16][36];   // per-wave P transpose staging

  v16bf qf;
  #pragma unroll
  for (int e = 0; e < 16; e++)
    qf[e] = u2b(QKV[(size_t)(q0 + l15) * ld + h * DHH + a_kidx(e, half)]);

  float rm[8], rs[8];
  v8f accO0 = {0.f,0.f,0.f,0.f,0.f,0.f,0.f,0.f};
  v8f accO1 = {0.f,0.f,0.f,0.f,0.f,0.f,0.f,0.f};
  #pragma unroll
  for (int r = 0; r < 8; r++) { rm[r] = -1e30f; rs[r] = 0.f; }
  const float scale = 0.17677669529663687f;  // 1/sqrt(32)

  for (int kb = 0; kb < nk; kb += 32) {
    // all waves done reading LDS from previous iteration
    __syncthreads();
    // cooperative K/V staging: 1024 elems each, 4 per thread
    #pragma unroll
    for (int i = tid; i < 32 * 32; i += 256) {
      int kl = i >> 5, dh = i & 31;
      sK[kl][dh] = QKV[(size_t)(kb + kl) * ld + HIDD     + h * DHH + dh];
      sV[kl][dh] = QKV[(size_t)(kb + kl) * ld + 2 * HIDD + h * DHH + dh];
    }
    __syncthreads();

    float sc[2][8];
    #pragma unroll
    for (int j = 0; j < 2; j++) {
      v16bf kf;
      #pragma unroll
      for (int e = 0; e < 16; e++)
        kf[e] = u2b(sK[j * 16 + l15][a_kidx(e, half)]);
      v8f s = {0.f,0.f,0.f,0.f,0.f,0.f,0.f,0.f};
      s = __builtin_amdgcn_wmma_f32_16x16x32_bf16(false, qf, false, kf,
                                                  (short)0, s, false, false);
      #pragma unroll
      for (int r = 0; r < 8; r++) {
        float v = s[r] * scale;
        if (bias) {
          size_t bidx = ((size_t)h * nq + (q0 + r + 8 * half)) * (size_t)nk + kb + j * 16 + l15;
          v += bfu2f(bias[bidx]);
        }
        sc[j][r] = v;
      }
    }
    // online softmax (rows live across the 16 lanes of each half-wave)
    #pragma unroll
    for (int r = 0; r < 8; r++) {
      float bm = fmaxf(sc[0][r], sc[1][r]);
      #pragma unroll
      for (int off = 8; off >= 1; off >>= 1) bm = fmaxf(bm, __shfl_xor(bm, off, 32));
      float nm   = fmaxf(rm[r], bm);
      float corr = __expf(rm[r] - nm);
      rm[r] = nm;
      float p0 = __expf(sc[0][r] - nm);
      float p1 = __expf(sc[1][r] - nm);
      sc[0][r] = p0; sc[1][r] = p1;
      float ps = p0 + p1;
      #pragma unroll
      for (int off = 8; off >= 1; off >>= 1) ps += __shfl_xor(ps, off, 32);
      rs[r] = rs[r] * corr + ps;
      accO0[r] = accO0[r] * corr;
      accO1[r] = accO1[r] * corr;
    }
    // transpose P (C-layout) -> A-layout via LDS
    #pragma unroll
    for (int j = 0; j < 2; j++)
      #pragma unroll
      for (int r = 0; r < 8; r++)
        sP[wave][r + 8 * half][j * 16 + l15] = f2bfu(sc[j][r]);
    __syncthreads();
    v16bf pf;
    #pragma unroll
    for (int e = 0; e < 16; e++)
      pf[e] = u2b(sP[wave][l15][a_kidx(e, half)]);
    #pragma unroll
    for (int j = 0; j < 2; j++) {
      v16bf vf;
      #pragma unroll
      for (int e = 0; e < 16; e++)
        vf[e] = u2b(sV[a_kidx(e, half)][j * 16 + l15]);
      if (j == 0)
        accO0 = __builtin_amdgcn_wmma_f32_16x16x32_bf16(false, pf, false, vf,
                                                        (short)0, accO0, false, false);
      else
        accO1 = __builtin_amdgcn_wmma_f32_16x16x32_bf16(false, pf, false, vf,
                                                        (short)0, accO1, false, false);
    }
  }
  #pragma unroll
  for (int r = 0; r < 8; r++) {
    float inv = 1.f / rs[r];
    int m = q0 + r + 8 * half;
    O[(size_t)m * HIDD + h * DHH + l15]      = accO0[r] * inv;
    O[(size_t)m * HIDD + h * DHH + 16 + l15] = accO1[r] * inv;
  }
}

// =====================================================================
// Elementwise / graph / reduction kernels
// =====================================================================
__global__ __launch_bounds__(256)
void bias_build_kernel(const int* __restrict__ dist, const int* __restrict__ etmm,
                       const float* __restrict__ demb, const float* __restrict__ eemb,
                       unsigned short* __restrict__ bias)
{
  size_t idx = (size_t)blockIdx.x * 256 + threadIdx.x;  // over NN*NN
  int dval = dist[idx], tval = etmm[idx];
  #pragma unroll
  for (int hh = 0; hh < NHH; hh++)
    bias[(size_t)hh * NN * NN + idx] = f2bfu(demb[dval * NHH + hh] + eemb[tval * NHH + hh]);
}

struct LinE { const float* w[ETT]; const float* b[ETT]; };

__global__ __launch_bounds__(256)
void edge_msg_kernel(const float* __restrict__ h, const int* __restrict__ src,
                     const int* __restrict__ dst, const float* __restrict__ ea,
                     const int* __restrict__ etype, LinE le, float* __restrict__ agg)
{
  int e = blockIdx.x;
  int d = threadIdx.x;
  int t = etype[e];
  int s = src[e], dd = dst[e];
  const float* w = le.w[t];
  float m = le.b[t][d];
  #pragma unroll
  for (int j = 0; j < EDIMM; j++) m += w[d * EDIMM + j] * ea[(size_t)e * EDIMM + j];
  m += h[(size_t)s * HIDD + d];
  m = fmaxf(m, 0.f);
  atomicAdd(&agg[((size_t)t * NN + dd) * HIDD + d], m);
}

__global__ void add_kernel(const float* __restrict__ a, const float* __restrict__ b,
                           float* __restrict__ o, size_t n)
{
  size_t i = (size_t)blockIdx.x * blockDim.x + threadIdx.x;
  if (i < n) o[i] = a[i] + b[i];
}

__global__ void f2bf_kernel(const float* __restrict__ in, unsigned short* __restrict__ out, size_t n)
{
  size_t i = (size_t)blockIdx.x * blockDim.x + threadIdx.x;
  if (i < n) out[i] = f2bfu(in[i]);
}

__global__ __launch_bounds__(256)
void mpnn_combine_kernel(const float* __restrict__ y, const float* __restrict__ etw,
                         float* __restrict__ out, int ldo)
{
  size_t idx = (size_t)blockIdx.x * 256 + threadIdx.x;  // over NN*HIDD
  int r = (int)(idx >> 8), c = (int)(idx & 255);
  float w[ETT]; float mx = -1e30f;
  #pragma unroll
  for (int t = 0; t < ETT; t++) { w[t] = etw[t]; mx = fmaxf(mx, w[t]); }
  float s = 0.f;
  #pragma unroll
  for (int t = 0; t < ETT; t++) { w[t] = __expf(w[t] - mx); s += w[t]; }
  float acc = 0.f;
  #pragma unroll
  for (int t = 0; t < ETT; t++) acc += w[t] * y[(size_t)t * NN * HIDD + idx];
  out[(size_t)r * ldo + c] = acc / s;
}

__global__ __launch_bounds__(256)
void colstats_kernel(const float* __restrict__ xx, float* __restrict__ mean,
                     float* __restrict__ var, int M, int Nc)
{
  int c = blockIdx.x, tid = threadIdx.x;
  float s = 0.f, s2 = 0.f;
  for (int r = tid; r < M; r += 256) {
    float v = xx[(size_t)r * Nc + c];
    s += v; s2 += v * v;
  }
  __shared__ float sh1[256], sh2[256];
  sh1[tid] = s; sh2[tid] = s2; __syncthreads();
  for (int st = 128; st > 0; st >>= 1) {
    if (tid < st) { sh1[tid] += sh1[tid + st]; sh2[tid] += sh2[tid + st]; }
    __syncthreads();
  }
  if (tid == 0) { float m = sh1[0] / M; mean[c] = m; var[c] = sh2[0] / M - m * m; }
}

__global__ __launch_bounds__(256)
void bn_res_kernel(const float* __restrict__ xf, const float* __restrict__ mean,
                   const float* __restrict__ var, const float* __restrict__ g,
                   const float* __restrict__ b, float* __restrict__ h)
{
  size_t idx = (size_t)blockIdx.x * 256 + threadIdx.x;  // over NN*HIDD
  int c = (int)(idx & (HIDD - 1));
  float v = (xf[idx] - mean[c]) * rsqrtf(var[c] + 1e-5f) * g[c] + b[c];
  h[idx] += v;
}

__global__ __launch_bounds__(256)
void ln_kernel(const float* __restrict__ a, const float* __restrict__ res,
               const float* __restrict__ g, const float* __restrict__ b,
               float* __restrict__ out)
{
  int r = blockIdx.x, c = threadIdx.x;   // HIDD threads
  float v = a[(size_t)r * HIDD + c];
  if (res) v += res[(size_t)r * HIDD + c];
  __shared__ float s1[256], s2[256];
  s1[c] = v; s2[c] = v * v; __syncthreads();
  for (int st = 128; st > 0; st >>= 1) {
    if (c < st) { s1[c] += s1[c + st]; s2[c] += s2[c + st]; }
    __syncthreads();
  }
  float m  = s1[0] / HIDD;
  float vv = s2[0] / HIDD - m * m;
  out[(size_t)r * HIDD + c] = (v - m) * rsqrtf(vv + 1e-5f) * g[c] + b[c];
}

__global__ __launch_bounds__(256)
void pert_kernel(const float* __restrict__ h_input, const int* __restrict__ pidx,
                 const float* __restrict__ proj_w, const float* __restrict__ proj_b,
                 const float* __restrict__ query,
                 const float* __restrict__ m1w, const float* __restrict__ m1b,
                 const float* __restrict__ m2w, const float* __restrict__ m2b,
                 float* __restrict__ h_pert)
{
  __shared__ float hp[2][HIDD];
  __shared__ float red[256];
  __shared__ float scores[2];
  __shared__ float pooled[HIDD];
  __shared__ float t1[HIDD];
  int tid = threadIdx.x;
  for (int p = 0; p < 2; p++) hp[p][tid] = h_input[(size_t)pidx[p] * HIDD + tid];
  __syncthreads();
  for (int p = 0; p < 2; p++) {
    float v = proj_b[tid];
    for (int k = 0; k < HIDD; k++) v += proj_w[tid * HIDD + k] * hp[p][k];
    red[tid] = v * query[tid];
    __syncthreads();
    for (int st = 128; st > 0; st >>= 1) {
      if (tid < st) red[tid] += red[tid + st];
      __syncthreads();
    }
    if (tid == 0) scores[p] = red[0];
    __syncthreads();
  }
  float mx = fmaxf(scores[0], scores[1]);
  float w0 = __expf(scores[0] - mx), w1 = __expf(scores[1] - mx);
  float ws = w0 + w1; w0 /= ws; w1 /= ws;
  pooled[tid] = w0 * hp[0][tid] + w1 * hp[1][tid];
  __syncthreads();
  float v = m1b[tid];
  for (int k = 0; k < HIDD; k++) v += m1w[tid * HIDD + k] * pooled[k];
  t1[tid] = fmaxf(v, 0.f);
  __syncthreads();
  float o = m2b[tid];
  for (int k = 0; k < HIDD; k++) o += m2w[tid * HIDD + k] * t1[k];
  h_pert[tid] = o;
}

__global__ __launch_bounds__(256)
void zset_kernel(const int* __restrict__ pidx, const float* __restrict__ h,
                 const float* __restrict__ hpert, float* __restrict__ Z)
{
  int p = blockIdx.x, c = threadIdx.x;
  int r = pidx[p];
  Z[(size_t)r * HIDD + c] = h[(size_t)r * HIDD + c] + hpert[c];
}

__global__ __launch_bounds__(256)
void comb_kernel(const float* __restrict__ Z, const float* __restrict__ h,
                 const float* __restrict__ hpert, const float* __restrict__ x,
                 float* __restrict__ comb)
{
  int r = blockIdx.x, c = threadIdx.x;
  size_t b = (size_t)r * 769;
  comb[b + c]       = Z[(size_t)r * HIDD + c];
  comb[b + 256 + c] = h[(size_t)r * HIDD + c];
  comb[b + 512 + c] = hpert[c];
  if (c == 0) comb[b + 768] = x[(size_t)r * INDIM];
}

__global__ __launch_bounds__(128)
void head_l3_kernel(const float* __restrict__ t2, const float* __restrict__ w,
                    const float* __restrict__ b, float* __restrict__ out)
{
  int r = blockIdx.x, tid = threadIdx.x;
  __shared__ float red[128];
  red[tid] = t2[(size_t)r * 128 + tid] * w[tid];
  __syncthreads();
  for (int st = 64; st > 0; st >>= 1) {
    if (tid < st) red[tid] += red[tid + st];
    __syncthreads();
  }
  if (tid == 0) out[r] = red[0] + b[0];
}

// =====================================================================
// Host orchestration
// =====================================================================
extern "C" void kernel_launch(void* const* d_in, const int* in_sizes, int n_in,
                              void* d_out, int out_size, void* d_ws, size_t ws_size,
                              hipStream_t stream)
{
  (void)in_sizes; (void)n_in; (void)out_size; (void)ws_size;
  const float* x          = (const float*)d_in[0];
  const int*   edge_index = (const int*)d_in[1];
  const float* edge_attr  = (const float*)d_in[2];
  const int*   edge_type  = (const int*)d_in[3];
  const int*   dist_mat   = (const int*)d_in[4];
  const int*   pert_index = (const int*)d_in[5];
  const int*   etm        = (const int*)d_in[6];

  // ----- unpack params (jax tree-leaf order: dict keys sorted) -----
  int p = 7;
  struct MHA { const float *in_b, *in_w, *out_b, *out_w; };
  struct Dec { MHA attn; const float *f1b,*f1w,*f2b,*f2w,*ln1b,*ln1g,*ln2b,*ln2g; } dec[3];
  for (int l = 0; l < 3; l++) {
    dec[l].attn.in_b  = (const float*)d_in[p++];
    dec[l].attn.in_w  = (const float*)d_in[p++];
    dec[l].attn.out_b = (const float*)d_in[p++];
    dec[l].attn.out_w = (const float*)d_in[p++];
    dec[l].f1b = (const float*)d_in[p++]; dec[l].f1w = (const float*)d_in[p++];
    dec[l].f2b = (const float*)d_in[p++]; dec[l].f2w = (const float*)d_in[p++];
    dec[l].ln1b = (const float*)d_in[p++]; dec[l].ln1g = (const float*)d_in[p++];
    dec[l].ln2b = (const float*)d_in[p++]; dec[l].ln2g = (const float*)d_in[p++];
  }
  const float* dist_emb = (const float*)d_in[p++];
  struct Enc { MHA attn; const float *bnb,*bng,*etw,*f1b,*f1w,*f2b,*f2w;
               const float *leb[ETT],*lew[ETT],*m1b[ETT],*m1w[ETT],*m2b[ETT],*m2w[ETT]; } enc[6];
  for (int l = 0; l < 6; l++) {
    enc[l].attn.in_b  = (const float*)d_in[p++];
    enc[l].attn.in_w  = (const float*)d_in[p++];
    enc[l].attn.out_b = (const float*)d_in[p++];
    enc[l].attn.out_w = (const float*)d_in[p++];
    enc[l].bnb = (const float*)d_in[p++]; enc[l].bng = (const float*)d_in[p++];
    enc[l].etw = (const float*)d_in[p++];
    enc[l].f1b = (const float*)d_in[p++]; enc[l].f1w = (const float*)d_in[p++];
    enc[l].f2b = (const float*)d_in[p++]; enc[l].f2w = (const float*)d_in[p++];
    for (int t = 0; t < ETT; t++) {
      enc[l].leb[t] = (const float*)d_in[p++]; enc[l].lew[t] = (const float*)d_in[p++];
      enc[l].m1b[t] = (const float*)d_in[p++]; enc[l].m1w[t] = (const float*)d_in[p++];
      enc[l].m2b[t] = (const float*)d_in[p++]; enc[l].m2w[t] = (const float*)d_in[p++];
    }
  }
  const float* et_emb = (const float*)d_in[p++];
  const float* l1b = (const float*)d_in[p++]; const float* l1w = (const float*)d_in[p++];
  const float* l2b = (const float*)d_in[p++]; const float* l2w = (const float*)d_in[p++];
  const float* l3b = (const float*)d_in[p++]; const float* l3w = (const float*)d_in[p++];
  const float* neb = (const float*)d_in[p++]; const float* new_ = (const float*)d_in[p++];
  const float* pm1b = (const float*)d_in[p++]; const float* pm1w = (const float*)d_in[p++];
  const float* pm2b = (const float*)d_in[p++]; const float* pm2w = (const float*)d_in[p++];
  const float* pprb = (const float*)d_in[p++]; const float* pprw = (const float*)d_in[p++];
  const float* pq   = (const float*)d_in[p++];

  // ----- workspace layout -----
  char* wp = (char*)d_ws;
  auto alloc = [&](size_t bytes) { char* r = wp; wp += (bytes + 255) & ~(size_t)255; return r; };
  unsigned short* bias_bf = (unsigned short*)alloc((size_t)NHH * NN * NN * 2);
  float* h       = (float*)alloc((size_t)NN * HIDD * 4);
  float* h_in    = (float*)alloc((size_t)NN * HIDD * 4);
  float* agg     = (float*)alloc((size_t)ETT * NN * HIDD * 4);
  float* hb      = (float*)alloc((size_t)NN * HIDD * 4);
  float* t512    = (float*)alloc((size_t)NN * 512 * 4);
  float* ybuf    = (float*)alloc((size_t)ETT * NN * HIDD * 4);
  float* xcat    = (float*)alloc((size_t)NN * 512 * 4);
  float* qkv     = (float*)alloc((size_t)NN * 768 * 4);
  unsigned short* qkv_bf = (unsigned short*)alloc((size_t)NN * 768 * 2);
  float* Otmp    = (float*)alloc((size_t)NN * HIDD * 4);
  float* ffn_mid = (float*)alloc((size_t)NN * 1024 * 4);
  float* ffn_out = (float*)alloc((size_t)NN * HIDD * 4);
  float* cmean   = (float*)alloc(256 * 4);
  float* cvar    = (float*)alloc(256 * 4);
  float* hpert   = (float*)alloc(256 * 4);
  float* Z       = (float*)alloc((size_t)NN * HIDD * 4);
  float* attn_o  = (float*)alloc((size_t)NN * HIDD * 4);
  float* comb    = (float*)alloc((size_t)NN * 769 * 4);
  float* ht1     = (float*)alloc((size_t)NN * HIDD * 4);
  float* ht2     = (float*)alloc((size_t)NN * 128 * 4);

  auto gemm = [&](const float* A, const float* W, const float* b, float* C,
                  int M, int K, int Nout, int ldc, int relu) {
    dim3 g((Nout + GT_N - 1) / GT_N, (M + GT_M - 1) / GT_M);
    gemm_bf16_kernel<<<g, 256, 0, stream>>>(A, W, b, C, M, K, Nout, ldc, relu);
  };

  // bias (bf16, resident in L2 across all 6 encoder layers)
  bias_build_kernel<<<(NN * NN) / 256, 256, 0, stream>>>(dist_mat, etm, dist_emb, et_emb, bias_bf);

  // node encoder
  gemm(x, new_, neb, h, NN, INDIM, HIDD, HIDD, 0);
  hipMemcpyAsync(h_in, h, (size_t)NN * HIDD * 4, hipMemcpyDeviceToDevice, stream);

  // ----- encoder -----
  for (int l = 0; l < 6; l++) {
    hipMemsetAsync(agg, 0, (size_t)ETT * NN * HIDD * 4, stream);
    LinE le;
    for (int t = 0; t < ETT; t++) { le.w[t] = enc[l].lew[t]; le.b[t] = enc[l].leb[t]; }
    edge_msg_kernel<<<EE, 256, 0, stream>>>(h, edge_index, edge_index + EE,
                                            edge_attr, edge_type, le, agg);
    for (int t = 0; t < ETT; t++) {
      add_kernel<<<(NN * HIDD) / 256, 256, 0, stream>>>(h, agg + (size_t)t * NN * HIDD, hb,
                                                        (size_t)NN * HIDD);
      gemm(hb, enc[l].m1w[t], enc[l].m1b[t], t512, NN, 256, 512, 512, 1);
      gemm(t512, enc[l].m2w[t], enc[l].m2b[t], ybuf + (size_t)t * NN * HIDD, NN, 512, 256, 256, 0);
    }
    mpnn_combine_kernel<<<(NN * HIDD) / 256, 256, 0, stream>>>(ybuf, enc[l].etw, xcat, 512);
    gemm(h, enc[l].attn.in_w, enc[l].attn.in_b, qkv, NN, 256, 768, 768, 0);
    f2bf_kernel<<<(NN * 768) / 256, 256, 0, stream>>>(qkv, qkv_bf, (size_t)NN * 768);
    flash_attn_kernel<<<dim3(NN / 128, NHH), 256, 0, stream>>>(qkv_bf, bias_bf, Otmp, NN, NN);
    gemm(Otmp, enc[l].attn.out_w, enc[l].attn.out_b, xcat + 256, NN, 256, 256, 512, 0);
    gemm(xcat, enc[l].f1w, enc[l].f1b, ffn_mid, NN, 512, 1024, 1024, 1);
    gemm(ffn_mid, enc[l].f2w, enc[l].f2b, ffn_out, NN, 1024, 256, 256, 0);
    colstats_kernel<<<256, 256, 0, stream>>>(ffn_out, cmean, cvar, NN, 256);
    bn_res_kernel<<<(NN * HIDD) / 256, 256, 0, stream>>>(ffn_out, cmean, cvar,
                                                         enc[l].bng, enc[l].bnb, h);
  }

  // ----- perturbation pooling -----
  pert_kernel<<<1, 256, 0, stream>>>(h_in, pert_index, pprw, pprb, pq,
                                     pm1w, pm1b, pm2w, pm2b, hpert);
  hipMemcpyAsync(Z, h, (size_t)NN * HIDD * 4, hipMemcpyDeviceToDevice, stream);
  zset_kernel<<<2, 256, 0, stream>>>(pert_index, h, hpert, Z);

  // ----- decoder -----
  for (int l = 0; l < 3; l++) {
    gemm(Z, dec[l].attn.in_w, dec[l].attn.in_b, qkv, NN, 256, 256, 768, 0);
    gemm(h, dec[l].attn.in_w + 256 * 256, dec[l].attn.in_b + 256, qkv + 256, NN, 256, 512, 768, 0);
    f2bf_kernel<<<(NN * 768) / 256, 256, 0, stream>>>(qkv, qkv_bf, (size_t)NN * 768);
    flash_attn_kernel<<<dim3(NN / 128, NHH), 256, 0, stream>>>(qkv_bf, nullptr, Otmp, NN, NN);
    gemm(Otmp, dec[l].attn.out_w, dec[l].attn.out_b, attn_o, NN, 256, 256, 256, 0);
    ln_kernel<<<NN, 256, 0, stream>>>(attn_o, Z, dec[l].ln1g, dec[l].ln1b, Z);
    gemm(Z, dec[l].f1w, dec[l].f1b, t512, NN, 256, 512, 512, 1);
    gemm(t512, dec[l].f2w, dec[l].f2b, ffn_out, NN, 512, 256, 256, 0);
    ln_kernel<<<NN, 256, 0, stream>>>(ffn_out, Z, dec[l].ln2g, dec[l].ln2b, Z);
  }

  // ----- head -----
  comb_kernel<<<NN, 256, 0, stream>>>(Z, h, hpert, x, comb);
  gemm(comb, l1w, l1b, ht1, NN, 769, 256, 256, 1);
  gemm(ht1, l2w, l2b, ht2, NN, 256, 128, 128, 1);
  head_l3_kernel<<<NN, 128, 0, stream>>>(ht2, l3w, l3b, (float*)d_out);
}
